// PNANet_13365938226037
// MI455X (gfx1250) — matrix operand order repaired and checked by reading, not compiled, not run on hardware
//
#include <hip/hip_runtime.h>
#include <hip/hip_fp16.h>

#define NN 20000
#define NE 320000
#define NG 128
#define IND 16
#define HD 64
#define NL 4
#define NC 10
#define AVG_D_LOG 2.8332133440562162f

typedef __attribute__((ext_vector_type(16))) _Float16 v16h;
typedef __attribute__((ext_vector_type(8)))  _Float16 v8h;
typedef __attribute__((ext_vector_type(8)))  float    v8f;

union V16 { v16h v; v8h h[2]; };

typedef __attribute__((address_space(3))) void lds_void;

__device__ __forceinline__ unsigned ldsOffset(const void* p) {
  return (unsigned)(size_t)(lds_void*)(void*)(size_t)p;
}

__device__ __forceinline__ unsigned flipF(float f) {
  unsigned u = __float_as_uint(f);
  return (u & 0x80000000u) ? ~u : (u | 0x80000000u);
}
__device__ __forceinline__ float unflipF(unsigned u) {
  return __uint_as_float((u & 0x80000000u) ? (u ^ 0x80000000u) : ~u);
}
__device__ __forceinline__ void atomAddF(float* p, float v) {
  __hip_atomic_fetch_add(p, v, __ATOMIC_RELAXED, __HIP_MEMORY_SCOPE_AGENT);
}
__device__ __forceinline__ void atomMaxU(unsigned* p, unsigned v) {
  __hip_atomic_fetch_max(p, v, __ATOMIC_RELAXED, __HIP_MEMORY_SCOPE_AGENT);
}
__device__ __forceinline__ void atomMinU(unsigned* p, unsigned v) {
  __hip_atomic_fetch_min(p, v, __ATOMIC_RELAXED, __HIP_MEMORY_SCOPE_AGENT);
}

// ---------------------------------------------------------------- degree
__global__ void k_degree(const int* __restrict__ dst, float* __restrict__ deg) {
  int e = blockIdx.x * blockDim.x + threadIdx.x;
  if (e < NE) atomAddF(&deg[dst[e]], 1.0f);
}

// ---------------------------------------------------------------- scalers
__global__ void k_scalers(const float* __restrict__ deg, float* __restrict__ amp,
                          float* __restrict__ att, float* __restrict__ invd,
                          float* __restrict__ hasnb) {
  int n = blockIdx.x * blockDim.x + threadIdx.x;
  if (n >= NN) return;
  float d = deg[n];
  float logd = __logf(d + 1.0f);
  amp[n]   = logd * (1.0f / AVG_D_LOG);
  att[n]   = AVG_D_LOG / fmaxf(logd, 1e-5f);
  invd[n]  = 1.0f / fmaxf(d, 1.0f);
  hasnb[n] = (d > 0.0f) ? 1.0f : 0.0f;
}

// ---------------------------------------------------------------- preW -> f16, transposed to [l][n][k]
__global__ void k_prew16(const float* __restrict__ preW, _Float16* __restrict__ w16) {
  int i = blockIdx.x * blockDim.x + threadIdx.x;
  if (i >= NL * HD * 2 * HD) return;
  int l = i >> 13;          // 8192 elements per layer
  int rem = i & 8191;
  int n = rem >> 7, k = rem & 127;
  w16[i] = (_Float16)preW[l * 8192 + k * HD + n];
}

// ---------------------------------------------------------------- embedding
__global__ void k_embed(const float* __restrict__ h, const float* __restrict__ embW,
                        const float* __restrict__ embB, float* __restrict__ x,
                        _Float16* __restrict__ xh) {
  int i = blockIdx.x * blockDim.x + threadIdx.x;
  if (i >= NN * HD) return;
  int n = i >> 6, c = i & 63;
  float acc = embB[c];
#pragma unroll
  for (int k = 0; k < IND; ++k) acc = fmaf(h[n * IND + k], embW[k * HD + c], acc);
  x[i] = acc;
  xh[i] = (_Float16)acc;
}

// ---------------------------------------------------------------- agg init
__global__ void k_agg_init(float* __restrict__ sum, float* __restrict__ sumsq,
                           unsigned* __restrict__ mx, unsigned* __restrict__ mn) {
  int i = blockIdx.x * blockDim.x + threadIdx.x;
  if (i >= NN * HD) return;
  sum[i] = 0.0f;
  sumsq[i] = 0.0f;
  mx[i] = 0x007FFFFFu;  // flip(-inf)
  mn[i] = 0xFF800000u;  // flip(+inf)
}

// ---------------------------------------------------------------- edge pretrans + fused multi-agg
// One wave = 32 edges (two M=16 tiles sharing each B fragment) x 64 output cols.
// K = 128 (src||dst halves of x). A per ISA 16-bit 16x32 layout.
// Weights staged into LDS via GLOBAL_LOAD_ASYNC_TO_LDS_B128 (ASYNCcnt path).
__global__ void __launch_bounds__(256)
k_edge(const _Float16* __restrict__ xh, const _Float16* __restrict__ w16_l,
       const float* __restrict__ preB_l, const int* __restrict__ src,
       const int* __restrict__ dst, float* __restrict__ sum,
       float* __restrict__ sumsq, unsigned* __restrict__ mx,
       unsigned* __restrict__ mn) {
  __shared__ _Float16 Wt[HD * 2 * HD];  // [n][k], 64 x 128 f16 = 16 KB
  int tid = threadIdx.x;

  // --- async stage of the 16 KB weight tile: 256 threads x 4 x 16 B ---
  {
    unsigned ldsbase = ldsOffset(&Wt[0]);
    const char* gsrc = (const char*)w16_l;
#pragma unroll
    for (int it = 0; it < 4; ++it) {
      unsigned boff = (unsigned)(it * 256 + tid) * 16u;
      unsigned ldsoff = ldsbase + boff;
      unsigned long long ga = (unsigned long long)(size_t)(gsrc + boff);
      asm volatile("global_load_async_to_lds_b128 %0, %1, off"
                   :: "v"(ldsoff), "v"(ga) : "memory");
    }
    asm volatile("s_wait_asynccnt 0x0" ::: "memory");
  }
  __syncthreads();

  int wave = tid >> 5, lane = tid & 31;
  int e0 = (blockIdx.x * 8 + wave) * 32;
  if (e0 >= NE) return;

  int r = lane & 15;       // edge-row (A) / out-col (B) within tile
  int kh = lane >> 4;      // K-half selector per ISA layout
  const _Float16* srow0 = xh + (long)src[e0 + r] * HD;
  const _Float16* drow0 = xh + (long)dst[e0 + r] * HD;
  const _Float16* srow1 = xh + (long)src[e0 + 16 + r] * HD;
  const _Float16* drow1 = xh + (long)dst[e0 + 16 + r] * HD;

  v8f acc0[4] = {v8f{}, v8f{}, v8f{}, v8f{}};
  v8f acc1[4] = {v8f{}, v8f{}, v8f{}, v8f{}};

#pragma unroll
  for (int kc = 0; kc < 4; ++kc) {           // four K=32 chunks (2 src, 2 dst)
    const _Float16* rp0 = (kc < 2) ? srow0 : drow0;
    const _Float16* rp1 = (kc < 2) ? srow1 : drow1;
    int kb = (kc & 1) * 32;
    V16 a0, a1;
    a0.h[0] = *(const v8h*)(rp0 + kb + kh * 8);
    a0.h[1] = *(const v8h*)(rp0 + kb + 16 + kh * 8);
    a1.h[0] = *(const v8h*)(rp1 + kb + kh * 8);
    a1.h[1] = *(const v8h*)(rp1 + kb + 16 + kh * 8);
#pragma unroll
    for (int t = 0; t < 4; ++t) {            // four N=16 output tiles
      const _Float16* wp = &Wt[(t * 16 + r) * 128 + kc * 32 + kh * 8];
      V16 b;
      b.h[0] = *(const v8h*)(wp);
      b.h[1] = *(const v8h*)(wp + 16);
      acc0[t] = __builtin_amdgcn_wmma_f32_16x16x32_f16(
          false, a0.v, false, b.v, (short)0, acc0[t], false, false);
      acc1[t] = __builtin_amdgcn_wmma_f32_16x16x32_f16(
          false, a1.v, false, b.v, (short)0, acc1[t], false, false);
    }
  }

  // C/D layout: VGPR g, lane<16 -> row g, col lane; lane>=16 -> row g+8, col lane-16
  int rowoff = kh * 8;
#pragma unroll
  for (int half = 0; half < 2; ++half) {
    const v8f* acc = half ? acc1 : acc0;
    int eb = e0 + half * 16;
#pragma unroll
    for (int g = 0; g < 8; ++g) {
      int e = eb + g + rowoff;
      long dbase = (long)dst[e] * HD;
#pragma unroll
      for (int t = 0; t < 4; ++t) {
        int col = t * 16 + r;
        float m = acc[t][g] + preB_l[col];
        long idx = dbase + col;
        atomAddF(&sum[idx], m);
        atomAddF(&sumsq[idx], m * m);
        unsigned fm = flipF(m);
        atomMaxU(&mx[idx], fm);
        atomMinU(&mn[idx], fm);
      }
    }
  }
}

// ---------------------------------------------------------------- posttrans + mix + residual
__global__ void __launch_bounds__(64)
k_post(const float* __restrict__ sum, const float* __restrict__ sumsq,
       const unsigned* __restrict__ mx, const unsigned* __restrict__ mn,
       const float* __restrict__ amp, const float* __restrict__ att,
       const float* __restrict__ invd, const float* __restrict__ hasnb,
       const float* __restrict__ snorm, const float* __restrict__ postW_l,
       const float* __restrict__ postB_l, const float* __restrict__ mixW_l,
       const float* __restrict__ mixB_l, float* __restrict__ x,
       _Float16* __restrict__ xh) {
  int n = blockIdx.x, j = threadIdx.x;
  __shared__ float v[13 * HD];
  __shared__ float yb[HD];
  long base = (long)n * HD + j;
  float iv = invd[n], am = amp[n], at = att[n], hn = hasnb[n];
  float mean = sum[base] * iv;
  float msq  = sumsq[base] * iv;
  float sd   = sqrtf(fmaxf(msq - mean * mean, 0.0f) + 1e-5f);
  float mxv  = (hn > 0.0f) ? unflipF(mx[base]) : 0.0f;
  float mnv  = (hn > 0.0f) ? unflipF(mn[base]) : 0.0f;
  float xv   = x[base];
  float a4[4] = {mean, mxv, mnv, sd};
#pragma unroll
  for (int b = 0; b < 4; ++b) {
    v[b * 64 + j]       = a4[b];
    v[256 + b * 64 + j] = a4[b] * am;
    v[512 + b * 64 + j] = a4[b] * at;
  }
  v[768 + j] = xv;
  __syncthreads();
  float acc = postB_l[j];
#pragma unroll 4
  for (int k = 0; k < 13 * HD; ++k) acc = fmaf(v[k], postW_l[k * HD + j], acc);
  yb[j] = acc;
  __syncthreads();
  float acc2 = mixB_l[j];
#pragma unroll 8
  for (int k = 0; k < HD; ++k) acc2 = fmaf(yb[k], mixW_l[k * HD + j], acc2);
  acc2 = (acc2 > 0.0f) ? acc2 : 0.01f * acc2;  // leaky_relu
  acc2 *= snorm[n];
  float xn = xv + acc2;
  x[base] = xn;
  xh[base] = (_Float16)xn;
}

// ---------------------------------------------------------------- graph sum
__global__ void k_gsum(const float* __restrict__ x, const int* __restrict__ gid,
                       float* __restrict__ gsum, float* __restrict__ gcnt) {
  int i = blockIdx.x * blockDim.x + threadIdx.x;
  if (i >= NN * HD) return;
  int n = i >> 6, c = i & 63;
  int g = gid[n];
  atomAddF(&gsum[g * HD + c], x[i]);
  if (c == 0) atomAddF(&gcnt[g], 1.0f);
}

// ---------------------------------------------------------------- readout MLP
__global__ void __launch_bounds__(64)
k_readout(const float* __restrict__ gsum, const float* __restrict__ gcnt,
          const float* __restrict__ r1W, const float* __restrict__ r1B,
          const float* __restrict__ r2W, const float* __restrict__ r2B,
          const float* __restrict__ r3W, const float* __restrict__ r3B,
          float* __restrict__ out) {
  int g = blockIdx.x, j = threadIdx.x;
  __shared__ float hg[64], h1[32], h2[16];
  float cnt = fmaxf(gcnt[g], 1.0f);
  hg[j] = gsum[g * HD + j] / cnt;
  __syncthreads();
  if (j < 32) {
    float a = r1B[j];
#pragma unroll 8
    for (int k = 0; k < 64; ++k) a = fmaf(hg[k], r1W[k * 32 + j], a);
    h1[j] = fmaxf(a, 0.0f);
  }
  __syncthreads();
  if (j < 16) {
    float a = r2B[j];
#pragma unroll
    for (int k = 0; k < 32; ++k) a = fmaf(h1[k], r2W[k * 16 + j], a);
    h2[j] = fmaxf(a, 0.0f);
  }
  __syncthreads();
  if (j < NC) {
    float a = r3B[j];
#pragma unroll
    for (int k = 0; k < 16; ++k) a = fmaf(h2[k], r3W[k * NC + j], a);
    out[g * NC + j] = a;
  }
}

// ---------------------------------------------------------------- host
extern "C" void kernel_launch(void* const* d_in, const int* in_sizes, int n_in,
                              void* d_out, int out_size, void* d_ws, size_t ws_size,
                              hipStream_t stream) {
  const float* h     = (const float*)d_in[0];
  const float* snorm = (const float*)d_in[1];
  const float* embW  = (const float*)d_in[2];
  const float* embB  = (const float*)d_in[3];
  const float* preW  = (const float*)d_in[4];
  const float* preB  = (const float*)d_in[5];
  const float* postW = (const float*)d_in[6];
  const float* postB = (const float*)d_in[7];
  const float* mixW  = (const float*)d_in[8];
  const float* mixB  = (const float*)d_in[9];
  const float* r1W   = (const float*)d_in[10];
  const float* r1B   = (const float*)d_in[11];
  const float* r2W   = (const float*)d_in[12];
  const float* r2B   = (const float*)d_in[13];
  const float* r3W   = (const float*)d_in[14];
  const float* r3B   = (const float*)d_in[15];
  const int* src = (const int*)d_in[16];
  const int* dst = (const int*)d_in[17];
  const int* gid = (const int*)d_in[18];
  float* out = (float*)d_out;

  char* ws = (char*)d_ws;
  size_t off = 0;
  auto carve = [&](size_t bytes) -> void* {
    void* p = ws + off;
    off = (off + bytes + 255) & ~(size_t)255;
    return p;
  };
  float*    x     = (float*)carve((size_t)NN * HD * 4);
  _Float16* xh    = (_Float16*)carve((size_t)NN * HD * 2);
  float*    sum   = (float*)carve((size_t)NN * HD * 4);
  float*    sumsq = (float*)carve((size_t)NN * HD * 4);
  unsigned* mx    = (unsigned*)carve((size_t)NN * HD * 4);
  unsigned* mn    = (unsigned*)carve((size_t)NN * HD * 4);
  _Float16* w16   = (_Float16*)carve((size_t)NL * HD * 2 * HD * 2);
  float*    deg   = (float*)carve((size_t)NN * 4);
  float*    amp   = (float*)carve((size_t)NN * 4);
  float*    att   = (float*)carve((size_t)NN * 4);
  float*    invd  = (float*)carve((size_t)NN * 4);
  float*    hasnb = (float*)carve((size_t)NN * 4);
  float*    gsum  = (float*)carve((size_t)NG * HD * 4);
  float*    gcnt  = (float*)carve((size_t)NG * 4);

  hipMemsetAsync(deg, 0, (size_t)NN * 4, stream);
  k_degree<<<(NE + 255) / 256, 256, 0, stream>>>(dst, deg);
  k_scalers<<<(NN + 255) / 256, 256, 0, stream>>>(deg, amp, att, invd, hasnb);
  k_prew16<<<(NL * HD * 2 * HD + 255) / 256, 256, 0, stream>>>(preW, w16);
  k_embed<<<(NN * HD + 255) / 256, 256, 0, stream>>>(h, embW, embB, x, xh);

  for (int l = 0; l < NL; ++l) {
    k_agg_init<<<(NN * HD + 255) / 256, 256, 0, stream>>>(sum, sumsq, mx, mn);
    k_edge<<<NE / 256, 256, 0, stream>>>(xh, w16 + (size_t)l * 2 * HD * HD,
                                         preB + (size_t)l * HD, src, dst,
                                         sum, sumsq, mx, mn);
    k_post<<<NN, 64, 0, stream>>>(sum, sumsq, mx, mn, amp, att, invd, hasnb,
                                  snorm, postW + (size_t)l * 13 * HD * HD,
                                  postB + (size_t)l * HD,
                                  mixW + (size_t)l * HD * HD,
                                  mixB + (size_t)l * HD, x, xh);
  }

  hipMemsetAsync(gsum, 0, (size_t)NG * HD * 4, stream);
  hipMemsetAsync(gcnt, 0, (size_t)NG * 4, stream);
  k_gsum<<<(NN * HD + 255) / 256, 256, 0, stream>>>(x, gid, gsum, gcnt);
  k_readout<<<NG, 64, 0, stream>>>(gsum, gcnt, r1W, r1B, r2W, r2B, r3W, r3B, out);
}
